// Sample_Attention_LSTM_74440373174794
// MI455X (gfx1250) — compile-verified, matrix operand order checked by reading, not run on hardware
//
#include <hip/hip_runtime.h>
#include <cstdint>
#include <cstddef>

typedef __attribute__((ext_vector_type(16))) _Float16 v16h;
typedef __attribute__((ext_vector_type(8)))  _Float16 v8h;
typedef __attribute__((ext_vector_type(8)))  float    v8f;
typedef __attribute__((ext_vector_type(2)))  float    v2f;

#define B_    256
#define SEQ_  512
#define HOR_  32
#define NK_   8192
#define L1_   64
#define HID_  128
#define L2_   64
#define GATES_ 512   // 4*HID
#define HLEN_ 544    // SEQ + HORIZON

// ---------------------------------------------------------------------------
// GEMM1: distance[b][n] = dot(input[b][0:512], akey[n][0:512])  (f32 WMMA)
// one wave per 16x16 tile; grid (16 Mtiles, 512/8 Ntile-groups), 8 waves/block
// ---------------------------------------------------------------------------
__global__ __launch_bounds__(256) void gemm1_kernel(const float* __restrict__ inp,
                                                    const float* __restrict__ akey,
                                                    float* __restrict__ dist) {
  const int tid = threadIdx.x, lane = tid & 31, wave = tid >> 5;
  const int mt = blockIdx.x, nt = blockIdx.y * 8 + wave;
  const int m = lane & 15;
  const int kb = (lane & 16) ? 2 : 0;          // f32 A/B frag: lanes>=16 hold K=2,3
  const float* Ap = inp  + (size_t)(mt * 16 + m) * SEQ_ + kb;
  const float* Bp = akey + (size_t)(nt * 16 + m) * SEQ_ + kb;
  v8f acc0 = {}, acc1 = {};
  for (int k = 0; k < SEQ_; k += 8) {
    v2f a0 = *(const v2f*)(Ap + k);
    v2f b0 = *(const v2f*)(Bp + k);
    acc0 = __builtin_amdgcn_wmma_f32_16x16x4_f32(false, a0, false, b0, (short)0, acc0, false, false);
    v2f a1 = *(const v2f*)(Ap + k + 4);
    v2f b1 = *(const v2f*)(Bp + k + 4);
    acc1 = __builtin_amdgcn_wmma_f32_16x16x4_f32(false, a1, false, b1, (short)0, acc1, false, false);
  }
#pragma unroll
  for (int r = 0; r < 8; ++r) {
    int row = r + ((lane & 16) ? 8 : 0);
    dist[(size_t)(mt * 16 + row) * NK_ + nt * 16 + m] = acc0[r] + acc1[r];
  }
}

// ---------------------------------------------------------------------------
// Softmax with first-occurrence argmax zeroed (in place on dist)
// ---------------------------------------------------------------------------
__global__ __launch_bounds__(256) void softmax_kernel(float* __restrict__ dist) {
  __shared__ float sv[256];
  __shared__ int   si[256];
  const int b = blockIdx.x, tid = threadIdx.x;
  float* row = dist + (size_t)b * NK_;

  float m = -3.4e38f; int mi = NK_;
  for (int j = tid; j < NK_; j += 256) {
    float v = row[j];
    if (v > m || (v == m && j < mi)) { m = v; mi = j; }
  }
  sv[tid] = m; si[tid] = mi;
  __syncthreads();
  for (int s = 128; s > 0; s >>= 1) {
    if (tid < s) {
      float v2 = sv[tid + s]; int i2 = si[tid + s];
      if (v2 > sv[tid] || (v2 == sv[tid] && i2 < si[tid])) { sv[tid] = v2; si[tid] = i2; }
    }
    __syncthreads();
  }
  if (tid == 0) row[si[0]] = 0.0f;
  __syncthreads();

  float mm = -3.4e38f;
  for (int j = tid; j < NK_; j += 256) mm = fmaxf(mm, row[j]);
  sv[tid] = mm;
  __syncthreads();
  for (int s = 128; s > 0; s >>= 1) { if (tid < s) sv[tid] = fmaxf(sv[tid], sv[tid + s]); __syncthreads(); }
  const float M = sv[0];
  __syncthreads();

  float sum = 0.f;
  for (int j = tid; j < NK_; j += 256) sum += __expf(row[j] - M);
  sv[tid] = sum;
  __syncthreads();
  for (int s = 128; s > 0; s >>= 1) { if (tid < s) sv[tid] += sv[tid + s]; __syncthreads(); }
  const float inv = 1.0f / sv[0];
  for (int j = tid; j < NK_; j += 256) row[j] = __expf(row[j] - M) * inv;
}

// ---------------------------------------------------------------------------
// GEMM2: hist = a @ aval  (256 x 544, K=8192)  (f32 WMMA)
// ---------------------------------------------------------------------------
__global__ __launch_bounds__(256) void gemm2_kernel(const float* __restrict__ a,
                                                    const float* __restrict__ aval,
                                                    float* __restrict__ hist) {
  const int tid = threadIdx.x, lane = tid & 31, wave = tid >> 5;
  const int mt = blockIdx.x, nt = blockIdx.y * 8 + wave;
  if (nt >= (HLEN_ / 16)) return;                 // 34 N-tiles, wave-uniform
  const int m = lane & 15;
  const int kb = (lane & 16) ? 2 : 0;
  const int n = nt * 16 + m;
  const float* Ap = a + (size_t)(mt * 16 + m) * NK_ + kb;
  const float* Bp = aval + (size_t)kb * HLEN_ + n;
  v8f acc0 = {}, acc1 = {};
  for (int k = 0; k < NK_; k += 8) {
    v2f av = *(const v2f*)(Ap + k);
    v2f bv; bv[0] = Bp[(size_t)k * HLEN_]; bv[1] = Bp[(size_t)(k + 1) * HLEN_];
    acc0 = __builtin_amdgcn_wmma_f32_16x16x4_f32(false, av, false, bv, (short)0, acc0, false, false);
    v2f av1 = *(const v2f*)(Ap + k + 4);
    v2f bv1; bv1[0] = Bp[(size_t)(k + 4) * HLEN_]; bv1[1] = Bp[(size_t)(k + 5) * HLEN_];
    acc1 = __builtin_amdgcn_wmma_f32_16x16x4_f32(false, av1, false, bv1, (short)0, acc1, false, false);
  }
#pragma unroll
  for (int r = 0; r < 8; ++r) {
    int row = r + ((lane & 16) ? 8 : 0);
    hist[(size_t)(mt * 16 + row) * HLEN_ + n] = acc0[r] + acc1[r];
  }
}

// ---------------------------------------------------------------------------
// Feature window + L1: x[b][t][o] = tanh(b1[o] + in[b][t]*W1[o][0]
//                                        + sum_j hist[b][t+j]*W1[o][1+j])
// Stored as f16 for WMMA consumption. One block per (b, 8 timesteps).
// ---------------------------------------------------------------------------
__global__ __launch_bounds__(256) void featl1_kernel(const float* __restrict__ inp,
                                                     const float* __restrict__ hist,
                                                     const float* __restrict__ W1,
                                                     const float* __restrict__ b1,
                                                     _Float16* __restrict__ xf) {
  __shared__ float w1s[L1_ * (HOR_ + 1)];
  __shared__ float b1s[L1_];
  __shared__ float hs[8 + HOR_ - 1];
  __shared__ float ins[8];
  const int b = blockIdx.x, t0 = blockIdx.y * 8, tid = threadIdx.x;
  for (int i = tid; i < L1_ * (HOR_ + 1); i += 256) w1s[i] = W1[i];
  if (tid < L1_) b1s[tid] = b1[tid];
  if (tid < 8 + HOR_ - 1) hs[tid] = hist[(size_t)b * HLEN_ + t0 + tid];
  if (tid < 8) ins[tid] = inp[(size_t)b * SEQ_ + t0 + tid];
  __syncthreads();
  const int o = tid & 63;
#pragma unroll
  for (int pass = 0; pass < 2; ++pass) {
    int tt = (tid >> 6) + pass * 4;
    float acc = b1s[o] + ins[tt] * w1s[o * 33];
#pragma unroll
    for (int j = 0; j < HOR_; ++j) acc += hs[tt + j] * w1s[o * 33 + 1 + j];
    xf[((size_t)b * SEQ_ + (t0 + tt)) * L1_ + o] = (_Float16)tanhf(acc);
  }
}

// ---------------------------------------------------------------------------
// Persistent LSTM layer. One workgroup per 16-row batch tile; 8 waves each
// own 64 gate columns (4 N-tiles). Weight B-fragments (f16) kept in VGPRs,
// h in LDS, gates combined cross-wave through LDS.
// ---------------------------------------------------------------------------
template <int KIN, bool STORE_SEQ>
__global__ __launch_bounds__(256, 1) void lstm_kernel(const _Float16* __restrict__ xin,
                                                      const float* __restrict__ Wih,
                                                      const float* __restrict__ Whh,
                                                      const float* __restrict__ bih,
                                                      const float* __restrict__ bhh,
                                                      _Float16* __restrict__ hseq,
                                                      float* __restrict__ hlast) {
  constexpr int KFX = KIN / 32;
  __shared__ float gates[16][GATES_];                 // 32 KB
  __shared__ __align__(16) _Float16 hbuf[16][HID_];   // 4 KB
  const int tid = threadIdx.x, lane = tid & 31, wave = tid >> 5;
  const int bm = blockIdx.x * 16;
  const int nbase = wave * 64;
  const int m = lane & 15;
  const int hi16 = (lane >> 4) & 1;
  const int kbh = hi16 * 8;

  // --- preload f16 B-fragments for Wih^T and Whh^T into registers ---
  v16h wihB[4][KFX], whhB[4][4];
  float biasv[4];
#pragma unroll
  for (int nt = 0; nt < 4; ++nt) {
    const int g = nbase + nt * 16 + m;                // gate column for this lane
    biasv[nt] = bih[g] + bhh[g];
#pragma unroll
    for (int kf = 0; kf < KFX; ++kf)
#pragma unroll
      for (int h = 0; h < 16; ++h) {
        int k = kf * 32 + ((h & 8) << 1) + kbh + (h & 7);
        wihB[nt][kf][h] = (_Float16)Wih[(size_t)g * KIN + k];
      }
#pragma unroll
    for (int kf = 0; kf < 4; ++kf)
#pragma unroll
      for (int h = 0; h < 16; ++h) {
        int k = kf * 32 + ((h & 8) << 1) + kbh + (h & 7);
        whhB[nt][kf][h] = (_Float16)Whh[(size_t)g * HID_ + k];
      }
  }

  // state: thread owns 8 consecutive hidden units of one row
  const int prow = tid >> 4;
  const int pj = (tid & 15) * 8;
  float c[8];
#pragma unroll
  for (int i = 0; i < 8; ++i) { c[i] = 0.f; hbuf[prow][pj + i] = (_Float16)0.f; }
  __syncthreads();

  for (int t = 0; t < SEQ_; ++t) {
    // A-fragments: x_t from global (f16), h from LDS
    v16h xA[KFX], hA[4];
#pragma unroll
    for (int kf = 0; kf < KFX; ++kf) {
      const _Float16* p = xin + ((size_t)(bm + m) * SEQ_ + t) * KIN + kf * 32 + kbh;
      v8h lo = *(const v8h*)p;
      v8h hi = *(const v8h*)(p + 16);
#pragma unroll
      for (int i = 0; i < 8; ++i) { xA[kf][i] = lo[i]; xA[kf][i + 8] = hi[i]; }
    }
#pragma unroll
    for (int kf = 0; kf < 4; ++kf) {
      const _Float16* p = &hbuf[m][kf * 32 + kbh];
      v8h lo = *(const v8h*)p;
      v8h hi = *(const v8h*)(p + 16);
#pragma unroll
      for (int i = 0; i < 8; ++i) { hA[kf][i] = lo[i]; hA[kf][i + 8] = hi[i]; }
    }
    // gates = bias + x_t @ Wih^T + h @ Whh^T   (WMMA f16->f32)
    v8f acc[4];
#pragma unroll
    for (int nt = 0; nt < 4; ++nt)
#pragma unroll
      for (int r = 0; r < 8; ++r) acc[nt][r] = biasv[nt];
#pragma unroll
    for (int nt = 0; nt < 4; ++nt) {
#pragma unroll
      for (int kf = 0; kf < KFX; ++kf)
        acc[nt] = __builtin_amdgcn_wmma_f32_16x16x32_f16(false, xA[kf], false, wihB[nt][kf],
                                                         (short)0, acc[nt], false, false);
#pragma unroll
      for (int kf = 0; kf < 4; ++kf)
        acc[nt] = __builtin_amdgcn_wmma_f32_16x16x32_f16(false, hA[kf], false, whhB[nt][kf],
                                                         (short)0, acc[nt], false, false);
    }
    // activations -> gates LDS (gate type uniform per wave: i,f,o sigmoid; g tanh)
#pragma unroll
    for (int nt = 0; nt < 4; ++nt) {
      const int g = nbase + nt * 16 + m;
      const bool isg = (g >> 7) == 2;
#pragma unroll
      for (int r = 0; r < 8; ++r) {
        int row = r + hi16 * 8;
        float v = acc[nt][r];
        gates[row][g] = isg ? tanhf(v) : 1.0f / (1.0f + __expf(-v));
      }
    }
    __syncthreads();
    // state update
    v8h hpk;
#pragma unroll
    for (int i = 0; i < 8; ++i) {
      int j = pj + i;
      float iv = gates[prow][j];
      float fv = gates[prow][HID_ + j];
      float gv = gates[prow][2 * HID_ + j];
      float ov = gates[prow][3 * HID_ + j];
      c[i] = fv * c[i] + iv * gv;
      float hv = ov * tanhf(c[i]);
      hpk[i] = (_Float16)hv;
      if (!STORE_SEQ && t == SEQ_ - 1) hlast[(size_t)(bm + prow) * HID_ + j] = hv;
    }
#pragma unroll
    for (int i = 0; i < 8; ++i) hbuf[prow][pj + i] = hpk[i];
    if (STORE_SEQ)
      *(v8h*)(hseq + ((size_t)(bm + prow) * SEQ_ + t) * HID_ + pj) = hpk;
    __syncthreads();
  }
}

// ---------------------------------------------------------------------------
// Head: last = tanh(h1_last); l2 = tanh(last@W2^T+b2); out = l2@W3^T+b3
// ---------------------------------------------------------------------------
__global__ __launch_bounds__(128) void head_kernel(const float* __restrict__ h1last,
                                                   const float* __restrict__ W2,
                                                   const float* __restrict__ b2,
                                                   const float* __restrict__ W3,
                                                   const float* __restrict__ b3,
                                                   float* __restrict__ out) {
  __shared__ float lastS[HID_];
  __shared__ float l2S[L2_];
  const int b = blockIdx.x, tid = threadIdx.x;
  lastS[tid] = tanhf(h1last[(size_t)b * HID_ + tid]);
  __syncthreads();
  if (tid < L2_) {
    float acc = b2[tid];
    for (int k = 0; k < HID_; ++k) acc += lastS[k] * W2[tid * HID_ + k];
    l2S[tid] = tanhf(acc);
  }
  __syncthreads();
  if (tid < HOR_) {
    float acc = b3[tid];
    for (int k = 0; k < L2_; ++k) acc += l2S[k] * W3[tid * L2_ + k];
    out[(size_t)b * HOR_ + tid] = acc;
  }
}

// ---------------------------------------------------------------------------
extern "C" void kernel_launch(void* const* d_in, const int* in_sizes, int n_in,
                              void* d_out, int out_size, void* d_ws, size_t ws_size,
                              hipStream_t stream) {
  (void)in_sizes; (void)n_in; (void)out_size; (void)ws_size;
  const float* inp  = (const float*)d_in[0];
  const float* akey = (const float*)d_in[1];
  const float* aval = (const float*)d_in[2];
  const float* W1   = (const float*)d_in[3];
  const float* b1   = (const float*)d_in[4];
  const float* Wih0 = (const float*)d_in[5];
  const float* Whh0 = (const float*)d_in[6];
  const float* bih0 = (const float*)d_in[7];
  const float* bhh0 = (const float*)d_in[8];
  const float* Wih1 = (const float*)d_in[9];
  const float* Whh1 = (const float*)d_in[10];
  const float* bih1 = (const float*)d_in[11];
  const float* bhh1 = (const float*)d_in[12];
  const float* W2   = (const float*)d_in[13];
  const float* b2   = (const float*)d_in[14];
  const float* W3   = (const float*)d_in[15];
  const float* b3   = (const float*)d_in[16];

  char* ws = (char*)d_ws;
  size_t off = 0;
  float*    dist   = (float*)(ws + off);    off += (size_t)B_ * NK_ * 4;          // 8 MB
  float*    hist   = (float*)(ws + off);    off += (size_t)B_ * HLEN_ * 4;        // 0.55 MB
  _Float16* xf     = (_Float16*)(ws + off); off += (size_t)B_ * SEQ_ * L1_ * 2;   // 16.8 MB
  _Float16* h0     = (_Float16*)(ws + off); off += (size_t)B_ * SEQ_ * HID_ * 2;  // 33.5 MB
  float*    h1last = (float*)(ws + off);    off += (size_t)B_ * HID_ * 4;

  gemm1_kernel<<<dim3(16, 64), 256, 0, stream>>>(inp, akey, dist);
  softmax_kernel<<<dim3(B_), 256, 0, stream>>>(dist);
  gemm2_kernel<<<dim3(16, 5), 256, 0, stream>>>(dist, aval, hist);
  featl1_kernel<<<dim3(B_, SEQ_ / 8), 256, 0, stream>>>(inp, hist, W1, b1, xf);
  lstm_kernel<L1_, true><<<dim3(B_ / 16), 256, 0, stream>>>(xf, Wih0, Whh0, bih0, bhh0, h0, nullptr);
  lstm_kernel<HID_, false><<<dim3(B_ / 16), 256, 0, stream>>>(h0, Wih1, Whh1, bih1, bhh1, nullptr, h1last);
  head_kernel<<<dim3(B_), 128, 0, stream>>>(h1last, W2, b2, W3, b3, (float*)d_out);
}
